// Calibrated_Spectral_Mixer_80255758893790
// MI455X (gfx1250) — compile-verified, matrix operand order checked by reading, not run on hardware
//
#include <hip/hip_runtime.h>
#include <hip/hip_bf16.h>

typedef __attribute__((ext_vector_type(16))) __bf16 v16bf;
typedef __attribute__((ext_vector_type(8)))  __bf16 v8bf;
typedef __attribute__((ext_vector_type(8)))  float  v8f;

#define HH    101
#define WW    31
#define NTOK  3131     // HH*WW
#define NPAD  3136     // padded row stride for fxT (16B-aligned vector loads)
#define NB    32       // batch
#define NROW  100192   // NB*NTOK
#define EPSLN 1e-5f

// ---- workspace layout (bytes, all 256-aligned) ----
#define OFF_XB   ((size_t)0)            // bf16 x           : 32*3131*256*2 = 51,298,304
#define OFF_WBT  ((size_t)51314688)     // bf16 wbT[9][1024][256]           =  4,718,592
#define OFF_OWB  ((size_t)56033280)     // bf16 out_w[256][512]             =    262,144
#define OFF_FXT  ((size_t)56295424)     // bf16 fxT[32*8][64][3136]         = 102,760,448
#define OFF_XM   ((size_t)159055872)    // bf16 xm[32*8][3131][64]          = 102,596,608
#define OFF_OUTX ((size_t)261685248)    // bf16 outx[32*3131][512]          = 102,596,608
#define OFF_ZERO ((size_t)364314624)    // 256 B of zeros (async OOB redirect)

__device__ __forceinline__ v8f zero8f() {
  v8f z;
#pragma unroll
  for (int i = 0; i < 8; ++i) z[i] = 0.0f;
  return z;
}

__device__ __forceinline__ v8bf zero8b() {
  v8bf z;
#pragma unroll
  for (int i = 0; i < 8; ++i) z[i] = (__bf16)0.0f;
  return z;
}

// Load one 16x32 A (or Bt) fragment from a row-major tile of bf16.
__device__ __forceinline__ v16bf ld_frag(const __bf16* base, int row0, int stride, int k0) {
  const int lane = (int)(threadIdx.x & 31);
  const __bf16* p = base + (row0 + (lane & 15)) * stride + k0 + ((lane >> 4) << 3);
  v16bf r;
  ((v8bf*)&r)[0] = *(const v8bf*)p;
  ((v8bf*)&r)[1] = *(const v8bf*)(p + 16);
  return r;
}

__device__ __forceinline__ v8f wmma_bf16(v16bf a, v16bf b, v8f c) {
  return __builtin_amdgcn_wmma_f32_16x16x32_bf16(false, a, false, b, (short)0, c, false, false);
}

// CDNA5 async memory->LDS copy (ASYNCcnt-tracked); 16B per lane.
__device__ __forceinline__ void async_ld_b128(uint32_t lds_addr, const void* gaddr) {
  asm volatile("global_load_async_to_lds_b128 %0, %1, off"
               :: "v"(lds_addr), "v"(gaddr)
               : "memory");
}
__device__ __forceinline__ void wait_async0() {
  asm volatile("s_wait_asynccnt 0x0" ::: "memory");
}
__device__ __forceinline__ uint32_t lds_addr_of(const void* p) {
  return (uint32_t)(uintptr_t)p;   // low 32 bits of generic LDS pointer == LDS byte offset
}

// ---------------- pack kernels ----------------
__global__ void k_pack_x(const float* __restrict__ x, __bf16* __restrict__ xb, int n) {
  int i = blockIdx.x * 256 + threadIdx.x;
  if (i < n) xb[i] = (__bf16)x[i];
}

// wbT[kyx][cout(1024)][cin(256)]; cout<512 -> conv_fx_w, else conv_x_w. w layout (512,256,3,3)
__global__ void k_pack_w(const float* __restrict__ wfx, const float* __restrict__ wx,
                         __bf16* __restrict__ wbT) {
  int idx = blockIdx.x * 256 + threadIdx.x;
  if (idx >= 9 * 1024 * 256) return;
  int kyx  = idx / (1024 * 256);
  int rem  = idx - kyx * (1024 * 256);
  int cout = rem >> 8;
  int cin  = rem & 255;
  int ky = kyx / 3, kx = kyx - ky * 3;
  float v = (cout < 512) ? wfx[((cout * 256 + cin) * 3 + ky) * 3 + kx]
                         : wx[(((cout - 512) * 256 + cin) * 3 + ky) * 3 + kx];
  wbT[idx] = (__bf16)v;
}

__global__ void k_pack_ow(const float* __restrict__ ow, __bf16* __restrict__ owb, int n) {
  int i = blockIdx.x * 256 + threadIdx.x;
  if (i < n) owb[i] = (__bf16)ow[i];
}

// ---------------- conv as implicit GEMM, async double-buffered ----------------
// grid: (25 m-tiles of 128, 16 cout-tiles of 64, 32 batches); 256 thr = 8 waves.
// Tile 128x64, K = 2304 in 72 steps of 32. Each wave owns a 32x32 output (4 WMMAs/step).
__global__ __launch_bounds__(256) void k_conv(const __bf16* __restrict__ xb,
                                              const __bf16* __restrict__ wbT,
                                              const float* __restrict__ bias_fx,
                                              const float* __restrict__ bias_x,
                                              const __bf16* __restrict__ zpad,
                                              __bf16* __restrict__ fxT,
                                              __bf16* __restrict__ xm) {
  __shared__ __attribute__((aligned(16))) __bf16 As[2][128 * 32];
  __shared__ __attribute__((aligned(16))) __bf16 Bs[2][64 * 32];
  const int tid = threadIdx.x, lane = tid & 31, wid = tid >> 5;
  const int wr = wid >> 1, wc = wid & 1;       // wave: rows wr*32..+31, cols wc*32..+31
  const int mt = blockIdx.x, ct = blockIdx.y, b = blockIdx.z;
  const int n0 = mt * 128, co0 = ct * 64;
  const int sel = co0 >> 9;                    // 0: fx conv, 1: x conv
  const int h = (co0 >> 6) & 7;

  // loader mapping: A chunks {tid, tid+256} of 512 (row = c>>2, q = c&3); B chunk tid of 256
  const int ra0 = tid >> 2, ra1 = ra0 + 64, qa = tid & 3;
  const int rbb = tid >> 2, qb = tid & 3;
  const int na0 = n0 + ra0, na1 = n0 + ra1;
  const int yy0 = na0 / WW, xx0 = na0 - yy0 * WW;
  const int yy1 = na1 / WW, xx1 = na1 - yy1 * WW;

  const uint32_t ldsA0[2] = { lds_addr_of(&As[0][(ra0 << 5) + (qa << 3)]),
                              lds_addr_of(&As[1][(ra0 << 5) + (qa << 3)]) };
  const uint32_t ldsA1[2] = { lds_addr_of(&As[0][(ra1 << 5) + (qa << 3)]),
                              lds_addr_of(&As[1][(ra1 << 5) + (qa << 3)]) };
  const uint32_t ldsB[2]  = { lds_addr_of(&Bs[0][(rbb << 5) + (qb << 3)]),
                              lds_addr_of(&Bs[1][(rbb << 5) + (qb << 3)]) };

  auto issue = [&](int kk, int p) {
    const int kyx = kk >> 3, cblk = kk & 7;
    const int ky = kyx / 3, kx = kyx - ky * 3;
    const int coffA = (cblk << 5) + (qa << 3);
    // A chunk 0
    {
      const int iy = yy0 + ky - 1, ix = xx0 + kx - 1;
      const bool ok = (na0 < NTOK) && ((unsigned)iy < (unsigned)HH) && ((unsigned)ix < (unsigned)WW);
      const __bf16* gp = ok ? (xb + ((size_t)((b * HH + iy) * WW + ix) << 8) + coffA) : zpad;
      async_ld_b128(ldsA0[p], gp);
    }
    // A chunk 1
    {
      const int iy = yy1 + ky - 1, ix = xx1 + kx - 1;
      const bool ok = (na1 < NTOK) && ((unsigned)iy < (unsigned)HH) && ((unsigned)ix < (unsigned)WW);
      const __bf16* gp = ok ? (xb + ((size_t)((b * HH + iy) * WW + ix) << 8) + coffA) : zpad;
      async_ld_b128(ldsA1[p], gp);
    }
    // B chunk
    {
      const __bf16* gp = wbT + ((size_t)((kyx << 10) + co0 + rbb) << 8) + (cblk << 5) + (qb << 3);
      async_ld_b128(ldsB[p], gp);
    }
  };

  v8f acc00 = zero8f(), acc01 = zero8f(), acc10 = zero8f(), acc11 = zero8f();

  issue(0, 0);   // prologue: tile 0 -> buffer 0

  for (int kk = 0; kk < 72; ++kk) {
    const int p = kk & 1;
    wait_async0();          // my async loads for buffer p (and any in flight) done
    __syncthreads();        // everyone's tile-p data visible; prev reads of buf 1-p done
    if (kk + 1 < 72) issue(kk + 1, p ^ 1);
    const v16bf a0 = ld_frag(As[p], wr * 32,      32, 0);
    const v16bf a1 = ld_frag(As[p], wr * 32 + 16, 32, 0);
    const v16bf b0 = ld_frag(Bs[p], wc * 32,      32, 0);
    const v16bf b1 = ld_frag(Bs[p], wc * 32 + 16, 32, 0);
    acc00 = wmma_bf16(a0, b0, acc00);
    acc01 = wmma_bf16(a0, b1, acc01);
    acc10 = wmma_bf16(a1, b0, acc10);
    acc11 = wmma_bf16(a1, b1, acc11);
  }

  const int bh = b * 8 + h;
#pragma unroll
  for (int rm = 0; rm < 2; ++rm) {
#pragma unroll
    for (int cn = 0; cn < 2; ++cn) {
      const v8f acc = rm ? (cn ? acc11 : acc10) : (cn ? acc01 : acc00);
      const int dl = wc * 32 + cn * 16 + (lane & 15);   // 0..63 within cout tile
      const float bias = (sel ? bias_x : bias_fx)[(co0 & 511) + dl];
      const int rb = n0 + wr * 32 + rm * 16 + ((lane >> 4) << 3);
#pragma unroll
      for (int i = 0; i < 8; ++i) {
        const int nn = rb + i;
        if (nn < NTOK) {
          const float v = acc[i] + bias;
          if (sel == 0)
            fxT[((size_t)bh * 64 + dl) * NPAD + nn] = (__bf16)v;
          else
            xm[(((size_t)bh * NTOK + nn) << 6) + dl] = (__bf16)v;
        }
      }
    }
  }
}

// ---------------- spectral mixer per (b,h): 256 blocks ----------------
__global__ __launch_bounds__(256) void k_spectral(const __bf16* __restrict__ xm,
                                                  const __bf16* __restrict__ fxT,
                                                  const float* __restrict__ gate_w,
                                                  const float* __restrict__ gate_b,
                                                  const float* __restrict__ temperature,
                                                  const float* __restrict__ ln_gamma,
                                                  const float* __restrict__ ln_beta,
                                                  const float* __restrict__ mlp_w,
                                                  const float* __restrict__ inver,
                                                  __bf16* __restrict__ outx) {
  __shared__ __attribute__((aligned(16))) __bf16 gw[64 * 64];    // Bt for logits
  __shared__ __attribute__((aligned(16))) __bf16 mw[64 * 64];    // Bt for mlp (mlp_w^T)
  __shared__ __attribute__((aligned(16))) __bf16 xt[64 * 64];    // x_mid tile / normalized spec
  __shared__ __attribute__((aligned(16))) __bf16 ft[64 * 64];    // fx tile (Bt: [c][n])
  __shared__ __attribute__((aligned(16))) __bf16 et[64 * 64];    // eigens (T in pass1, RM in pass2)
  __shared__ __attribute__((aligned(16))) __bf16 osT[64 * 64];   // out_spec^T (Bt for out_x)
  __shared__ __attribute__((aligned(16))) float fstage[64 * 64]; // logits / spec staging
  __shared__ __attribute__((aligned(16))) float red[514];

  const int tid = threadIdx.x, lane = tid & 31, wid = tid >> 5;
  const int wr = wid >> 1, wc = wid & 1;
  const int bh = blockIdx.x, bb = bh >> 3, h = bh & 7;
  const float temp = fminf(fmaxf(temperature[h], 0.1f), 5.0f);
  const float rtemp = 1.0f / temp;

  for (int i = tid; i < 4096; i += 256) {
    gw[i] = (__bf16)gate_w[i];                         // gw[g][c] = gate_w[g][c]
    mw[i] = (__bf16)mlp_w[((i & 63) << 6) + (i >> 6)]; // mw[o][i] = mlp_w[i][o]
  }
  const size_t xmBase = ((size_t)bh * NTOK) << 6;
  const size_t fxBase = (size_t)bh * 64 * NPAD;

  v8f sp0 = zero8f(), sp1 = zero8f();
  __syncthreads();

  const int rbw = wr * 16 + ((lane >> 4) << 3);      // accumulator row base within 64
  const int cb0 = wc * 32 + (lane & 15);             // accumulator col (first 16-wide)

  // ================= PASS 1: spec = eigens^T @ fx =================
  for (int t = 0; t < 49; ++t) {
    const int n0 = t * 64;
    {
      const int rr = tid >> 2, q = tid & 3;
      const int nn = n0 + rr;
      if (nn < NTOK) {
        const v8bf* src = (const v8bf*)(xm + xmBase + ((size_t)nn << 6) + (q << 4));
        *(v8bf*)(xt + (rr << 6) + (q << 4)) = src[0];
        *(v8bf*)(xt + (rr << 6) + (q << 4) + 8) = src[1];
      } else {
        *(v8bf*)(xt + (rr << 6) + (q << 4)) = zero8b();
        *(v8bf*)(xt + (rr << 6) + (q << 4) + 8) = zero8b();
      }
      // ft[c][nlocal]  (c = rr)
      if (n0 + (q << 4) + 16 <= NTOK) {
        const v8bf* src = (const v8bf*)(fxT + fxBase + (size_t)rr * NPAD + n0 + (q << 4));
        *(v8bf*)(ft + (rr << 6) + (q << 4)) = src[0];
        *(v8bf*)(ft + (rr << 6) + (q << 4) + 8) = src[1];
      } else {
        for (int k = 0; k < 16; ++k) {
          const int nn2 = n0 + (q << 4) + k;
          ft[(rr << 6) + (q << 4) + k] =
              (nn2 < NTOK) ? fxT[fxBase + (size_t)rr * NPAD + nn2] : (__bf16)0.0f;
        }
      }
    }
    __syncthreads();
    // logits = xt @ gw^T
    v8f l0 = zero8f(), l1 = zero8f();
#pragma unroll
    for (int ks = 0; ks < 64; ks += 32) {
      const v16bf a  = ld_frag(xt, wr * 16, 64, ks);
      const v16bf b0 = ld_frag(gw, wc * 32, 64, ks);
      const v16bf b1 = ld_frag(gw, wc * 32 + 16, 64, ks);
      l0 = wmma_bf16(a, b0, l0);
      l1 = wmma_bf16(a, b1, l1);
    }
#pragma unroll
    for (int i = 0; i < 8; ++i) {
      fstage[(rbw + i) * 64 + cb0] = l0[i];
      fstage[(rbw + i) * 64 + cb0 + 16] = l1[i];
    }
    __syncthreads();
    if (tid < 64) {
      const int nn = n0 + tid;
      const float* lrow = fstage + tid * 64;
      if (nn < NTOK) {
        float mx = -1e30f;
        for (int g = 0; g < 64; ++g) mx = fmaxf(mx, (lrow[g] + gate_b[g]) * rtemp);
        float s = 0.0f;
        for (int g = 0; g < 64; ++g) s += __expf((lrow[g] + gate_b[g]) * rtemp - mx);
        const float rs = 1.0f / s;
        for (int g = 0; g < 64; ++g) {
          const float eig = __expf((lrow[g] + gate_b[g]) * rtemp - mx) * rs * inver[nn * 64 + g];
          et[g * 64 + tid] = (__bf16)eig;            // transposed: et[g][nlocal]
        }
      } else {
        for (int g = 0; g < 64; ++g) et[g * 64 + tid] = (__bf16)0.0f;
      }
    }
    __syncthreads();
    // spec += et @ ft^T   (A rows = g, Bt rows = c, K = nlocal)
#pragma unroll
    for (int ks = 0; ks < 64; ks += 32) {
      const v16bf a  = ld_frag(et, wr * 16, 64, ks);
      const v16bf b0 = ld_frag(ft, wc * 32, 64, ks);
      const v16bf b1 = ld_frag(ft, wc * 32 + 16, 64, ks);
      sp0 = wmma_bf16(a, b0, sp0);
      sp1 = wmma_bf16(a, b1, sp1);
    }
    __syncthreads();
  }

  // ------- LayerNorm over all 4096 spec elements -------
#pragma unroll
  for (int i = 0; i < 8; ++i) {
    fstage[(rbw + i) * 64 + cb0] = sp0[i];
    fstage[(rbw + i) * 64 + cb0 + 16] = sp1[i];
  }
  __syncthreads();
  {
    float s = 0.0f, s2 = 0.0f;
    for (int k = 0; k < 16; ++k) {
      const float v = fstage[tid * 16 + k];
      s += v;
      s2 += v * v;
    }
    red[tid] = s;
    red[256 + tid] = s2;
  }
  __syncthreads();
  if (tid == 0) {
    float S = 0.0f, S2 = 0.0f;
    for (int i = 0; i < 256; ++i) { S += red[i]; S2 += red[256 + i]; }
    const float mu = S * (1.0f / 4096.0f);
    const float var = S2 * (1.0f / 4096.0f) - mu * mu;
    red[512] = mu;
    red[513] = rsqrtf(var + EPSLN);
  }
  __syncthreads();
  {
    const float mu = red[512], rstd = red[513];
    for (int k = 0; k < 16; ++k) {
      const int idx = tid * 16 + k;
      xt[idx] = (__bf16)((fstage[idx] - mu) * rstd * ln_gamma[idx] + ln_beta[idx]);
    }
  }
  __syncthreads();
  // ------- out_spec = spec_norm @ mlp_w ; store transposed -------
  {
    v8f m0 = zero8f(), m1 = zero8f();
#pragma unroll
    for (int ks = 0; ks < 64; ks += 32) {
      const v16bf a  = ld_frag(xt, wr * 16, 64, ks);
      const v16bf b0 = ld_frag(mw, wc * 32, 64, ks);
      const v16bf b1 = ld_frag(mw, wc * 32 + 16, 64, ks);
      m0 = wmma_bf16(a, b0, m0);
      m1 = wmma_bf16(a, b1, m1);
    }
#pragma unroll
    for (int j = 0; j < 2; ++j) {
      const v8f m = j ? m1 : m0;
      const int o = wc * 32 + j * 16 + (lane & 15);
#pragma unroll
      for (int i = 0; i < 8; ++i) osT[o * 64 + rbw + i] = (__bf16)m[i];
    }
  }
  __syncthreads();

  // ================= PASS 2: out_x = eigens @ out_spec =================
  for (int t = 0; t < 49; ++t) {
    const int n0 = t * 64;
    {
      const int rr = tid >> 2, q = tid & 3;
      const int nn = n0 + rr;
      if (nn < NTOK) {
        const v8bf* src = (const v8bf*)(xm + xmBase + ((size_t)nn << 6) + (q << 4));
        *(v8bf*)(xt + (rr << 6) + (q << 4)) = src[0];
        *(v8bf*)(xt + (rr << 6) + (q << 4) + 8) = src[1];
      } else {
        *(v8bf*)(xt + (rr << 6) + (q << 4)) = zero8b();
        *(v8bf*)(xt + (rr << 6) + (q << 4) + 8) = zero8b();
      }
    }
    __syncthreads();
    v8f l0 = zero8f(), l1 = zero8f();
#pragma unroll
    for (int ks = 0; ks < 64; ks += 32) {
      const v16bf a  = ld_frag(xt, wr * 16, 64, ks);
      const v16bf b0 = ld_frag(gw, wc * 32, 64, ks);
      const v16bf b1 = ld_frag(gw, wc * 32 + 16, 64, ks);
      l0 = wmma_bf16(a, b0, l0);
      l1 = wmma_bf16(a, b1, l1);
    }
#pragma unroll
    for (int i = 0; i < 8; ++i) {
      fstage[(rbw + i) * 64 + cb0] = l0[i];
      fstage[(rbw + i) * 64 + cb0 + 16] = l1[i];
    }
    __syncthreads();
    if (tid < 64) {
      const int nn = n0 + tid;
      const float* lrow = fstage + tid * 64;
      if (nn < NTOK) {
        float mx = -1e30f;
        for (int g = 0; g < 64; ++g) mx = fmaxf(mx, (lrow[g] + gate_b[g]) * rtemp);
        float s = 0.0f;
        for (int g = 0; g < 64; ++g) s += __expf((lrow[g] + gate_b[g]) * rtemp - mx);
        const float rs = 1.0f / s;
        for (int g = 0; g < 64; ++g) {
          const float eig = __expf((lrow[g] + gate_b[g]) * rtemp - mx) * rs * inver[nn * 64 + g];
          et[tid * 64 + g] = (__bf16)eig;            // row-major: et[nlocal][g]
        }
      } else {
        for (int g = 0; g < 64; ++g) et[tid * 64 + g] = (__bf16)0.0f;
      }
    }
    __syncthreads();
    v8f o0 = zero8f(), o1 = zero8f();
#pragma unroll
    for (int ks = 0; ks < 64; ks += 32) {
      const v16bf a  = ld_frag(et, wr * 16, 64, ks);
      const v16bf b0 = ld_frag(osT, wc * 32, 64, ks);
      const v16bf b1 = ld_frag(osT, wc * 32 + 16, 64, ks);
      o0 = wmma_bf16(a, b0, o0);
      o1 = wmma_bf16(a, b1, o1);
    }
#pragma unroll
    for (int j = 0; j < 2; ++j) {
      const v8f o = j ? o1 : o0;
      const int c = wc * 32 + j * 16 + (lane & 15);
#pragma unroll
      for (int i = 0; i < 8; ++i) {
        const int nn = n0 + rbw + i;
        if (nn < NTOK)
          outx[(((size_t)bb * NTOK + nn) << 9) + (h << 6) + c] = (__bf16)o[i];
      }
    }
    __syncthreads();
  }
}

// ---------------- final projection: (BN x 512) @ out_w^T + out_b ----------------
__global__ __launch_bounds__(256) void k_outproj(const __bf16* __restrict__ outx,
                                                 const __bf16* __restrict__ owb,
                                                 const float* __restrict__ out_b,
                                                 float* __restrict__ y) {
  __shared__ __attribute__((aligned(16))) __bf16 As[64 * 32];
  __shared__ __attribute__((aligned(16))) __bf16 Bs[64 * 32];
  const int tid = threadIdx.x, lane = tid & 31, wid = tid >> 5;
  const int wr = wid >> 1, wc = wid & 1;
  const int r0 = blockIdx.x * 64, c0 = blockIdx.y * 64;
  const int rr = tid >> 2, q = tid & 3;
  v8f acc0 = zero8f(), acc1 = zero8f();

  for (int ks = 0; ks < 16; ++ks) {
    v8bf av = zero8b();
    const int row = r0 + rr;
    if (row < NROW)
      av = *(const v8bf*)(outx + ((size_t)row << 9) + (ks << 5) + (q << 3));
    const v8bf bv = *(const v8bf*)(owb + ((size_t)(c0 + rr) << 9) + (ks << 5) + (q << 3));
    __syncthreads();
    *(v8bf*)(As + (rr << 5) + (q << 3)) = av;
    *(v8bf*)(Bs + (rr << 5) + (q << 3)) = bv;
    __syncthreads();
    const v16bf a  = ld_frag(As, wr * 16, 32, 0);
    const v16bf b0 = ld_frag(Bs, wc * 32, 32, 0);
    const v16bf b1 = ld_frag(Bs, wc * 32 + 16, 32, 0);
    acc0 = wmma_bf16(a, b0, acc0);
    acc1 = wmma_bf16(a, b1, acc1);
  }

  const int rb = r0 + wr * 16 + ((lane >> 4) << 3);
#pragma unroll
  for (int j = 0; j < 2; ++j) {
    const v8f acc = j ? acc1 : acc0;
    const int c = c0 + wc * 32 + j * 16 + (lane & 15);
    const float bias = out_b[c];
#pragma unroll
    for (int i = 0; i < 8; ++i) {
      const int row = rb + i;
      if (row < NROW) y[(size_t)row * 256 + c] = acc[i] + bias;
    }
  }
}

extern "C" void kernel_launch(void* const* d_in, const int* in_sizes, int n_in,
                              void* d_out, int out_size, void* d_ws, size_t ws_size,
                              hipStream_t stream) {
  (void)in_sizes; (void)n_in; (void)out_size; (void)ws_size;
  const float* x           = (const float*)d_in[0];
  const float* conv_fx_w   = (const float*)d_in[1];
  const float* conv_fx_b   = (const float*)d_in[2];
  const float* conv_x_w    = (const float*)d_in[3];
  const float* conv_x_b    = (const float*)d_in[4];
  const float* gate_w      = (const float*)d_in[5];
  const float* gate_b      = (const float*)d_in[6];
  const float* temperature = (const float*)d_in[7];
  const float* ln_gamma    = (const float*)d_in[8];
  const float* ln_beta     = (const float*)d_in[9];
  const float* mlp_w       = (const float*)d_in[10];
  const float* out_w       = (const float*)d_in[11];
  const float* out_b       = (const float*)d_in[12];
  const float* inver       = (const float*)d_in[13];

  char* ws = (char*)d_ws;
  __bf16* xb   = (__bf16*)(ws + OFF_XB);
  __bf16* wbT  = (__bf16*)(ws + OFF_WBT);
  __bf16* owb  = (__bf16*)(ws + OFF_OWB);
  __bf16* fxT  = (__bf16*)(ws + OFF_FXT);
  __bf16* xm   = (__bf16*)(ws + OFF_XM);
  __bf16* outx = (__bf16*)(ws + OFF_OUTX);
  __bf16* zpad = (__bf16*)(ws + OFF_ZERO);

  hipMemsetAsync(ws + OFF_ZERO, 0, 256, stream);

  const int nX = NB * NTOK * 256;   // 25,649,152
  k_pack_x<<<(nX + 255) / 256, 256, 0, stream>>>(x, xb, nX);
  k_pack_w<<<(9 * 1024 * 256 + 255) / 256, 256, 0, stream>>>(conv_fx_w, conv_x_w, wbT);
  k_pack_ow<<<(256 * 512 + 255) / 256, 256, 0, stream>>>(out_w, owb, 256 * 512);

  k_conv<<<dim3(25, 16, NB), 256, 0, stream>>>(xb, wbT, conv_fx_b, conv_x_b, zpad, fxT, xm);

  k_spectral<<<dim3(NB * 8), 256, 0, stream>>>(xm, fxT, gate_w, gate_b, temperature,
                                               ln_gamma, ln_beta, mlp_w, inver, outx);

  k_outproj<<<dim3((NROW + 63) / 64, 4), 256, 0, stream>>>(outx, owb, out_b, (float*)d_out);
}
